// WaveletKANLayer_64132451664288
// MI455X (gfx1250) — matrix operand order repaired and checked by reading, not compiled, not run on hardware
//
#include <hip/hip_runtime.h>
#include <cstdint>

// WaveletKAN layer as a fused basis-generation + bf16 WMMA GEMM for gfx1250.
//
// out[n,o] = sum_{i,k} W[o,i,k] * basis(x[n,i]; k) + bias[o]
//   == GEMM  M=8192 (tokens) x N=512 (out) x Kred=8192 (=512 feats * 16 wavelets)
// W is row-major [o][i][k] with k innermost -> flat W2[o][ik]: B needs no transpose.
//
// Roofline: 68.7 GFLOP GEMM vs ~50 MB minimal HBM traffic (~2.2 us @ 23.3 TB/s)
// -> compute bound -> bf16 WMMA (v_wmma_f32_16x16x32_bf16) with f32 accumulate.
//
// Pre-kernel: W f32 -> bf16 into d_ws (512*8192*2 = 8.4 MB of scratch).
// Main kernel: BM=128, BN=256, BK=32, 512 threads (16 wave32 waves).
//   - basis tile computed with v_cos_f32 / v_exp_f32, packed bf16 -> LDS
//   - W bf16 tile copied global->LDS with GLOBAL_LOAD_ASYNC_TO_LDS_B128 (ASYNCcnt)
//   - each wave: 2x4 tiles of v_wmma_f32_16x16x32_bf16
// Grid = (8192/128) x (512/256) = 128 workgroups; basis recompute factor = 2.

#define N_TOK 8192
#define IN_F  512
#define OUT_F 512
#define N_WAV 16
#define KRED  (IN_F * N_WAV)   // 8192

#define BM 128
#define BN 256
#define BK 32                  // 2 input features * 16 wavelets
#define THREADS 512            // 16 waves (wave32)
#define LDA 40                 // A row pitch in bf16 elements (32 + 8 pad -> 80B rows)
#define LDB 40                 // B row pitch in bf16 elements

typedef __attribute__((ext_vector_type(16))) __bf16 v16bf;
typedef __attribute__((ext_vector_type(8)))  float  v8f;
typedef __attribute__((ext_vector_type(4)))  int    v4i;

#if defined(__HIP_DEVICE_COMPILE__) && defined(__gfx1250__) && \
    __has_builtin(__builtin_amdgcn_global_load_async_to_lds_b128) && \
    __has_builtin(__builtin_amdgcn_s_wait_asynccnt)
#define WKAN_ASYNC 1
#else
#define WKAN_ASYNC 0
#endif

static __device__ __forceinline__ unsigned short f2bf(float f) {
  unsigned int u = __builtin_bit_cast(unsigned int, f);
  u += 0x7fffu + ((u >> 16) & 1u);          // round-to-nearest-even
  return (unsigned short)(u >> 16);
}

// ---------------------------------------------------------------------------
// Kernel 1: convert wavelet_weights f32 -> packed bf16 pairs in workspace.
// ---------------------------------------------------------------------------
__global__ void wkan_convert_w(const float* __restrict__ w,
                               unsigned int* __restrict__ wbf, int n_pairs) {
  int i = blockIdx.x * blockDim.x + threadIdx.x;
  if (i < n_pairs) {
    float2 f = ((const float2*)w)[i];
    wbf[i] = (unsigned int)f2bf(f.x) | ((unsigned int)f2bf(f.y) << 16);
  }
}

// ---------------------------------------------------------------------------
// Kernel 2: fused basis + GEMM.
// ---------------------------------------------------------------------------
__global__ __launch_bounds__(THREADS, 1)
void wkan_wmma_kernel(const float* __restrict__ x,
                      const unsigned short* __restrict__ wbf,  // [OUT_F][KRED] bf16
                      const float* __restrict__ scales,
                      const float* __restrict__ trans,
                      const float* __restrict__ bias,
                      float* __restrict__ out) {
  __shared__ unsigned short As[BM * LDA];  // basis tile (bf16), rows padded
  __shared__ unsigned short Bs[BN * LDB];  // weight tile (bf16), rows padded

  const int t    = (int)threadIdx.x;
  const int wave = t >> 5;
  const int lane = t & 31;
  const int half = lane >> 4;   // 0: lanes 0-15, 1: lanes 16-31
  const int l16  = lane & 15;

  const int m0 = (int)blockIdx.x * BM;
  const int o0 = (int)blockIdx.y * BN;

  // 4x4 wave grid; each wave owns a 32(M) x 64(N) sub-tile = 2x4 WMMA tiles.
  const int wm = wave >> 2;     // 0..3  -> M offset wm*32
  const int wn = wave & 3;      // 0..3  -> N offset wn*64

  // A-producer role: (row, which-of-2-features, which-8-wavelets)
  const int a_row = t & 127;
  const int a_ii  = (t >> 7) & 1;
  const int a_kh  = (t >> 8) & 1;
  // B-copier role: thread copies 16 bytes (8 bf16) of one W row per chunk
  const int b_row = t >> 1;     // 0..255
  const int b_j   = t & 1;

  // Per-thread wavelet constants for its 8 wavelets (static register indices).
  float isv[8], trv[8];
#pragma unroll
  for (int j = 0; j < 8; ++j) {
    int k = a_kh * 8 + j;
    isv[j] = 1.0f / (scales[k] + 1e-6f);
    trv[j] = trans[k];
  }

  v8f acc[2][4];
#pragma unroll
  for (int mt = 0; mt < 2; ++mt)
#pragma unroll
    for (int nt = 0; nt < 4; ++nt)
      acc[mt][nt] = (v8f){0.f, 0.f, 0.f, 0.f, 0.f, 0.f, 0.f, 0.f};

  const unsigned short* wrow = wbf + (size_t)(o0 + b_row) * KRED;

  for (int c = 0; c < KRED / BK; ++c) {
    // ---- produce A tile: 8 basis values -> one ds_store_b128 ----
    {
      float xv = x[(m0 + a_row) * IN_F + (2 * c + a_ii)];
      unsigned int pk[4];
#pragma unroll
      for (int j = 0; j < 4; ++j) {
        float t0 = (xv - trv[2 * j])     * isv[2 * j];
        float t1 = (xv - trv[2 * j + 1]) * isv[2 * j + 1];
        float b0 = __cosf(5.0f * t0) * __expf(-0.5f * t0 * t0);
        float b1 = __cosf(5.0f * t1) * __expf(-0.5f * t1 * t1);
        pk[j] = (unsigned int)f2bf(b0) | ((unsigned int)f2bf(b1) << 16);
      }
      *(uint4*)&As[a_row * LDA + a_ii * 16 + a_kh * 8] =
          make_uint4(pk[0], pk[1], pk[2], pk[3]);
    }
    // ---- copy B tile: 8 bf16 per thread, async global -> LDS ----
    {
      const unsigned short* gsrc = wrow + c * BK + b_j * 16;
      unsigned short*       ldst = &Bs[b_row * LDB + b_j * 16];
#if WKAN_ASYNC
      __builtin_amdgcn_global_load_async_to_lds_b128(
          (__attribute__((address_space(1))) v4i*)(uintptr_t)gsrc,
          (__attribute__((address_space(3))) v4i*)(unsigned int)(uintptr_t)ldst,
          0, 0);
#else
      *(uint4*)ldst = *(const uint4*)gsrc;
#endif
      __builtin_prefetch(gsrc + BK, 0, 1);   // next chunk -> global_prefetch_b8
    }
#if WKAN_ASYNC
    __builtin_amdgcn_s_wait_asynccnt(0);
#endif
    __syncthreads();

    // ---- B fragments (32x16 bf16): lane l16 = column o, K 8h..8h+7 then 16+8h.. ----
    v16bf bf[4];
#pragma unroll
    for (int nt = 0; nt < 4; ++nt) {
      const uint4* p = (const uint4*)&Bs[(wn * 64 + nt * 16 + l16) * LDB + half * 8];
      union { uint4 u[2]; v16bf v; } f;
      f.u[0] = p[0];
      f.u[1] = p[2];
      bf[nt] = f.v;
    }
    // ---- A fragments (16x32 bf16) + WMMA ----
#pragma unroll
    for (int mt = 0; mt < 2; ++mt) {
      const int arow = wm * 32 + mt * 16 + l16;
      union { uint4 u[2]; v16bf v; } fa;
      fa.u[0] = *(const uint4*)&As[arow * LDA + half * 8];        // K 0-7 / 8-15
      fa.u[1] = *(const uint4*)&As[arow * LDA + 16 + half * 8];   // K 16-23 / 24-31
      v16bf af = fa.v;
#pragma unroll
      for (int nt = 0; nt < 4; ++nt)
        acc[mt][nt] = __builtin_amdgcn_wmma_f32_16x16x32_bf16(
            false, af, false, bf[nt], (short)0, acc[mt][nt], false, false);
    }
    __syncthreads();
  }

  // ---- epilogue: bias + store (N across lanes -> coalesced f32 stores) ----
  float bv[4];
#pragma unroll
  for (int nt = 0; nt < 4; ++nt)
    bv[nt] = bias[o0 + wn * 64 + nt * 16 + l16];

#pragma unroll
  for (int mt = 0; mt < 2; ++mt)
#pragma unroll
    for (int nt = 0; nt < 4; ++nt) {
      const int col = o0 + wn * 64 + nt * 16 + l16;
#pragma unroll
      for (int r = 0; r < 8; ++r) {
        const int row = m0 + wm * 32 + mt * 16 + r + half * 8;  // ISA C/D layout
        out[row * OUT_F + col] = acc[mt][nt][r] + bv[nt];
      }
    }
}

// ---------------------------------------------------------------------------
extern "C" void kernel_launch(void* const* d_in, const int* in_sizes, int n_in,
                              void* d_out, int out_size, void* d_ws, size_t ws_size,
                              hipStream_t stream) {
  const float* x      = (const float*)d_in[0];
  const float* w      = (const float*)d_in[1];   // [512][512][16] f32
  const float* scales = (const float*)d_in[2];
  const float* trans  = (const float*)d_in[3];
  const float* bias   = (const float*)d_in[4];
  float* out = (float*)d_out;

  // Workspace: W as bf16, [OUT_F][KRED] = 512*8192*2 B = 8.4 MB.
  unsigned int* wbf = (unsigned int*)d_ws;

  const int n_pairs = OUT_F * KRED / 2;  // 2,097,152 packed bf16 pairs
  wkan_convert_w<<<dim3((n_pairs + 255) / 256), dim3(256), 0, stream>>>(w, wbf, n_pairs);

  wkan_wmma_kernel<<<dim3(N_TOK / BM, OUT_F / BN), dim3(THREADS), 0, stream>>>(
      x, (const unsigned short*)wbf, scales, trans, bias, out);
}
